// Refine_35450660061279
// MI455X (gfx1250) — matrix operand order repaired and checked by reading, not compiled, not run on hardware
//
#include <hip/hip_runtime.h>
#include <stdint.h>
#include <math.h>

// ---------------- problem constants (match reference) ----------------
constexpr int N_ENT = 20000, N_REL = 400, H = 256, T = 8, E = 50000, Bq = 2048;
constexpr int CH = 50, NH = 8, TOPK = 50, L = 2;
constexpr int MASKW = (N_ENT + 31) / 32;     // 625 u32 words per relation

typedef float  v8f  __attribute__((ext_vector_type(8)));
typedef __bf16 v16bf __attribute__((ext_vector_type(16)));

union AFrag { v16bf v; uint4 q[2]; };
union CFrag { v8f v; float f[8]; };

__device__ __forceinline__ unsigned short f2bf(float f) {
  union { float f; uint32_t u; } x; x.f = f;
  uint32_t r = x.u + 0x7fffu + ((x.u >> 16) & 1u);   // round-to-nearest-even
  return (unsigned short)(r >> 16);
}
__device__ __forceinline__ float bf2f(unsigned short b) {
  union { uint32_t u; float f; } x; x.u = ((uint32_t)b) << 16; return x.f;
}
__device__ __forceinline__ uint32_t pk2(float a, float b) {
  return (uint32_t)f2bf(a) | ((uint32_t)f2bf(b) << 16);
}

// =====================================================================
// Templated WMMA GEMM: C[M,N] = act( A[M,K] * B + addC + bias )
//  ABF16 : A is bf16 (else f32, converted while staging)
//  GATHER: A rows indirected through aRowIdx
//  BTRANS: B is NxK row-major (x @ W.T); else KxN row-major
//  CBF16 : C written as bf16 (else f32)
//  Block tile 64x64, 256 threads = 8 waves (4x2); each wave computes a
//  16x32 strip -> 2x v_wmma_f32_16x16x32_bf16 per 32-wide K step, A reused.
// =====================================================================
template <int ABF16, int GATHER, int BTRANS, int CBF16>
__global__ __launch_bounds__(256)
void gemm_wmma_kernel(const void* __restrict__ Ap, const int* __restrict__ aRowIdx,
                      const float* __restrict__ Bp,
                      const float* __restrict__ addC, const float* __restrict__ bias,
                      void* __restrict__ Cp, int M, int N, int K, int act) {
  __shared__ __align__(16) unsigned short lA[64 * 32];   // [m][k]
  __shared__ __align__(16) unsigned short lB[64 * 32];   // [n][k] (k-contiguous per n)
  const int tid  = threadIdx.x;
  const int lane = tid & 31, wid = tid >> 5;
  const int half = lane >> 4, lr = lane & 15;
  const int wm = wid >> 1, wn = wid & 1;                 // wave 4x2 grid
  const int m0 = blockIdx.y * 64, n0 = blockIdx.x * 64;

  // staging assignments (one 8-wide chunk per thread per tile)
  const int am = tid >> 2;            // 0..63 row within tile
  const int kc = (tid & 3) * 8;       // k chunk 0/8/16/24
  const int gmS = m0 + am;
  const bool mok = gmS < M;
  const long aRow = mok ? (GATHER ? (long)aRowIdx[gmS] : (long)gmS) : 0;
  const int gnS = n0 + am;            // for BTRANS staging (row of B = output col)
  const bool nokS = gnS < N;

  CFrag c0, c1;
  #pragma unroll
  for (int i = 0; i < 8; ++i) { c0.f[i] = 0.f; c1.f[i] = 0.f; }

  const int kSteps = K >> 5;
  for (int ks = 0; ks < kSteps; ++ks) {
    const int k0 = ks << 5;
    __syncthreads();
    // ---- stage A (64x32) ----
    {
      uint4 pack = {0u, 0u, 0u, 0u};
      if (mok) {
        if (ABF16) {
          pack = *(const uint4*)((const unsigned short*)Ap + aRow * (long)K + k0 + kc);
        } else {
          const float* ap = (const float*)Ap + aRow * (long)K + k0 + kc;
          float4 f0 = *(const float4*)ap;
          float4 f1 = *(const float4*)(ap + 4);
          pack.x = pk2(f0.x, f0.y); pack.y = pk2(f0.z, f0.w);
          pack.z = pk2(f1.x, f1.y); pack.w = pk2(f1.z, f1.w);
        }
      }
      *(uint4*)&lA[am * 32 + kc] = pack;
    }
    // ---- stage B (64 cols x 32 k) ----
    if (BTRANS) {
      uint4 pack = {0u, 0u, 0u, 0u};
      if (nokS) {
        const float* bp = Bp + (long)gnS * K + k0 + kc;
        float4 f0 = *(const float4*)bp;
        float4 f1 = *(const float4*)(bp + 4);
        pack.x = pk2(f0.x, f0.y); pack.y = pk2(f0.z, f0.w);
        pack.z = pk2(f1.x, f1.y); pack.w = pk2(f1.z, f1.w);
      }
      *(uint4*)&lB[am * 32 + kc] = pack;
    } else {
      #pragma unroll
      for (int i = 0; i < 8; ++i) {
        int idx = tid + (i << 8);
        int bn = idx >> 5, bk = idx & 31;
        int gn = n0 + bn;
        lB[bn * 32 + bk] = (gn < N) ? f2bf(Bp[(long)(k0 + bk) * N + gn]) : (unsigned short)0;
      }
    }
    __syncthreads();
    // ---- fragments (documented CDNA5 VGPR layouts) ----
    // A 16x32: lanes 0-15 (row lr): elems 0..7 = K0..7, 8..15 = K16..23;
    //          lanes 16-31:         elems 0..7 = K8..15, 8..15 = K24..31.
    AFrag a, b;
    const unsigned short* pa = &lA[(wm * 16 + lr) * 32];
    a.q[0] = *(const uint4*)(pa + half * 8);
    a.q[1] = *(const uint4*)(pa + 16 + half * 8);
    // B 32x16: same K order as A (lanes 0-15: K0..7,K16..23; lanes 16-31: K8..15,K24..31).
    const unsigned short* pb0 = &lB[(wn * 32 + lr) * 32];
    b.q[0] = *(const uint4*)(pb0 + half * 8);
    b.q[1] = *(const uint4*)(pb0 + 16 + half * 8);
    c0.v = __builtin_amdgcn_wmma_f32_16x16x32_bf16(false, a.v, false, b.v,
                                                   (short)0, c0.v, false, false);
    const unsigned short* pb1 = &lB[(wn * 32 + 16 + lr) * 32];
    b.q[0] = *(const uint4*)(pb1 + half * 8);
    b.q[1] = *(const uint4*)(pb1 + 16 + half * 8);
    c1.v = __builtin_amdgcn_wmma_f32_16x16x32_bf16(false, a.v, false, b.v,
                                                   (short)0, c1.v, false, false);
  }
  // ---- epilogue: elem i -> row = wm*16 + half*8 + i, col = lr of each subtile ----
  #pragma unroll
  for (int sub = 0; sub < 2; ++sub) {
    const CFrag& c = sub ? c1 : c0;
    const int gn = n0 + wn * 32 + sub * 16 + lr;
    if (gn >= N) continue;
    float bvv = bias ? bias[gn] : 0.f;
    #pragma unroll
    for (int i = 0; i < 8; ++i) {
      int gm = m0 + wm * 16 + half * 8 + i;
      if (gm >= M) continue;
      float v = c.f[i] + bvv;
      long o = (long)gm * N + gn;
      if (addC) v += addC[o];
      if (act == 1) v = fmaxf(v, 0.f);
      else if (act == 2) v = 1.f / (1.f + expf(-v));
      if (CBF16) ((unsigned short*)Cp)[o] = f2bf(v);
      else       ((float*)Cp)[o] = v;
    }
  }
}

// ---------------- helper kernels ----------------
__global__ void zero_kernel(float* p, long n) {
  long i = (long)blockIdx.x * 256 + threadIdx.x;
  if (i < n) p[i] = 0.f;
}

__global__ __launch_bounds__(256)
void rownorm_kernel(const float* __restrict__ src, float* __restrict__ dst) {
  __shared__ float red[256];
  int r = blockIdx.x, t = threadIdx.x;
  float v = src[(long)r * H + t];
  red[t] = v * v; __syncthreads();
  for (int s = 128; s > 0; s >>= 1) { if (t < s) red[t] += red[t + s]; __syncthreads(); }
  float n = fmaxf(sqrtf(red[0]), 1e-12f);
  dst[(long)r * H + t] = v / n;
}

__global__ void mask_set_kernel(const int* __restrict__ src, const int* __restrict__ dst,
                                const int* __restrict__ rid, unsigned* __restrict__ mask) {
  int e = blockIdx.x * 256 + threadIdx.x;
  if (e >= E) return;
  int r = rid[e], s = src[e], d = dst[e];
  atomicOr(&mask[r * MASKW + (s >> 5)], 1u << (s & 31));
  atomicOr(&mask[r * MASKW + (d >> 5)], 1u << (d & 31));
}

// mean of member-node features per relation (dedup via bitmask)
__global__ __launch_bounds__(256)
void rel_agg_kernel(const unsigned* __restrict__ mask, const float* __restrict__ ent,
                    float* __restrict__ relEnt) {
  __shared__ unsigned sm[MASKW];
  int r = blockIdx.x, t = threadIdx.x;
  for (int i = t; i < MASKW; i += 256) sm[i] = mask[r * MASKW + i];
  __syncthreads();
  float acc = 0.f; int cnt = 0;
  for (int w = 0; w < MASKW; ++w) {
    unsigned u = sm[w];
    cnt += __popc(u);
    while (u) {
      int bpos = __ffs(u) - 1; u &= u - 1;
      acc += ent[(long)((w << 5) + bpos) * H + t];
    }
  }
  relEnt[r * H + t] = cnt > 0 ? acc / (float)cnt : 0.f;
}

__global__ void gru_in_kernel(const float* relEnt, const float* relEmb, float* gruIn) {
  int r = blockIdx.x, t = threadIdx.x;
  gruIn[(long)r * (2 * H) + t]     = relEnt[r * H + t];
  gruIn[(long)r * (2 * H) + H + t] = relEmb[r * H + t];
}

// GRU gates + l2norm fused
__global__ __launch_bounds__(256)
void gru_fuse_kernel(const float* __restrict__ gi, const float* __restrict__ gh,
                     const float* __restrict__ hprev, float* __restrict__ hnext) {
  __shared__ float red[256];
  int r = blockIdx.x, t = threadIdx.x;
  long b3 = (long)r * 3 * H;
  float ir = gi[b3 + t], iz = gi[b3 + H + t], in = gi[b3 + 2 * H + t];
  float hr = gh[b3 + t], hz = gh[b3 + H + t], hn = gh[b3 + 2 * H + t];
  float rg = 1.f / (1.f + expf(-(ir + hr)));
  float z  = 1.f / (1.f + expf(-(iz + hz)));
  float nn = tanhf(in + rg * hn);
  float hp = hprev[(long)r * H + t];
  float h  = (1.f - z) * nn + z * hp;
  red[t] = h * h; __syncthreads();
  for (int s = 128; s > 0; s >>= 1) { if (t < s) red[t] += red[t + s]; __syncthreads(); }
  float norm = fmaxf(sqrtf(red[0]), 1e-12f);
  hnext[(long)r * H + t] = h / norm;
}

// edge scatter: agg[dst] += h[src] + nrel[rid]; cnt[dst] += 1
__global__ void scatter_kernel(const int* __restrict__ src, const int* __restrict__ dst,
                               const int* __restrict__ rid, const float* __restrict__ h,
                               const float* __restrict__ nrel, float* __restrict__ agg,
                               float* __restrict__ cnt) {
  long idx = (long)blockIdx.x * 256 + threadIdx.x;
  int e = (int)(idx >> 6);          // 64 threads/edge, 4 channels each
  if (e >= E) return;
  int c = (int)(idx & 63) * 4;
  int s = src[e], d = dst[e], r = rid[e];
  const float* hs = h + (long)s * H + c;
  const float* rf = nrel + (long)r * H + c;
  float* out = agg + (long)d * H + c;
  atomicAdd(out + 0, hs[0] + rf[0]);
  atomicAdd(out + 1, hs[1] + rf[1]);
  atomicAdd(out + 2, hs[2] + rf[2]);
  atomicAdd(out + 3, hs[3] + rf[3]);
  if (c == 0) atomicAdd(cnt + d, 1.f);
}

__global__ void agg_norm_kernel(float* agg, const float* cnt) {
  long i = (long)blockIdx.x * 256 + threadIdx.x;
  if (i >= (long)N_ENT * H) return;
  float cn = cnt[i >> 8];
  agg[i] = cn > 0.f ? agg[i] / cn : 0.f;
}

__global__ void ent_update_kernel(float* ent, const float* u, const float* h) {
  long i = (long)blockIdx.x * 256 + threadIdx.x;
  if (i >= (long)N_ENT * H) return;
  float e = ent[i]; ent[i] = e + u[i] * (h[i] - e);
}

__global__ void gather_kernel(const float* ent, const float* rele, const int* subj,
                              const int* rel, const int* obj, float* es, float* er,
                              float* eo, float* cat2) {
  int b = blockIdx.x, t = threadIdx.x;
  float vs = ent[(long)subj[b] * H + t];
  float vr = rele[(long)rel[b] * H + t];
  float vo = ent[(long)obj[b] * H + t];
  es[(long)b * H + t] = vs; er[(long)b * H + t] = vr; eo[(long)b * H + t] = vo;
  cat2[(long)b * 2 * H + t] = vs; cat2[(long)b * 2 * H + H + t] = vr;
}

// ConvTransE conv stage fused with relu, emits bf16 (B, CH*H) feature matrix
__global__ __launch_bounds__(256)
void conv_feat_kernel(const float* __restrict__ r0, const float* __restrict__ r1,
                      const float* __restrict__ cw, const float* __restrict__ cb,
                      unsigned short* __restrict__ Y) {
  __shared__ float s0[H + 2], s1[H + 2];
  int b = blockIdx.x, t = threadIdx.x;
  s0[t + 1] = r0[(long)b * H + t];
  s1[t + 1] = r1[(long)b * H + t];
  if (t == 0) { s0[0] = 0.f; s1[0] = 0.f; s0[H + 1] = 0.f; s1[H + 1] = 0.f; }
  __syncthreads();
  for (int c = 0; c < CH; ++c) {
    const float* w = cw + c * 6;
    float y = cb[c] + s0[t] * w[0] + s0[t + 1] * w[1] + s0[t + 2] * w[2]
                    + s1[t] * w[3] + s1[t + 1] * w[4] + s1[t + 2] * w[5];
    Y[(long)b * (CH * H) + c * H + t] = f2bf(fmaxf(y, 0.f));
  }
}

// per-row top-50 via iterative argmax over an LDS bf16 copy (tie -> lower index)
__global__ __launch_bounds__(256)
void topk_kernel(const float* __restrict__ logits, int* __restrict__ topk) {
  __shared__ unsigned short row[N_ENT];
  __shared__ float rv[256];
  __shared__ int   ri[256];
  int b = blockIdx.x, t = threadIdx.x;
  for (int i = t; i < N_ENT; i += 256) row[i] = f2bf(logits[(long)b * N_ENT + i]);
  __syncthreads();
  for (int j = 0; j < TOPK; ++j) {
    float bm = -INFINITY; int bi = N_ENT;
    for (int i = t; i < N_ENT; i += 256) {
      float v = bf2f(row[i]);
      if (v > bm || (v == bm && i < bi)) { bm = v; bi = i; }
    }
    rv[t] = bm; ri[t] = bi; __syncthreads();
    for (int s = 128; s > 0; s >>= 1) {
      if (t < s) {
        float v2 = rv[t + s]; int i2 = ri[t + s];
        if (v2 > rv[t] || (v2 == rv[t] && i2 < ri[t])) { rv[t] = v2; ri[t] = i2; }
      }
      __syncthreads();
    }
    if (t == 0) { topk[b * TOPK + j] = ri[0]; row[ri[0]] = f2bf(-INFINITY); }
    __syncthreads();
  }
}

// fused attention: wave h handles head h; scores via shfl, wave-local softmax
__global__ __launch_bounds__(256)
void attn_kernel(const float* __restrict__ q, const unsigned short* __restrict__ kh,
                 const unsigned short* __restrict__ vh, float* __restrict__ out) {
  __shared__ float sc[NH][TOPK];
  int b = blockIdx.x, t = threadIdx.x;
  int head = t >> 5, dim = t & 31;
  float qv = q[(long)b * H + t];
  const float scale = 0.17677669529663687f;   // 1/sqrt(32)
  for (int s = 0; s < TOPK; ++s) {
    float p = qv * bf2f(kh[((long)b * TOPK + s) * H + t]);
    for (int off = 16; off > 0; off >>= 1) p += __shfl_xor(p, off, 32);
    if (dim == 0) sc[head][s] = p * scale;
  }
  float m = -INFINITY;
  for (int s = dim; s < TOPK; s += 32) m = fmaxf(m, sc[head][s]);
  for (int off = 16; off > 0; off >>= 1) m = fmaxf(m, __shfl_xor(m, off, 32));
  float sum = 0.f;
  for (int s = dim; s < TOPK; s += 32) { float e2 = expf(sc[head][s] - m); sc[head][s] = e2; sum += e2; }
  for (int off = 16; off > 0; off >>= 1) sum += __shfl_xor(sum, off, 32);
  float acc = 0.f;
  for (int s = 0; s < TOPK; ++s) acc += sc[head][s] * bf2f(vh[((long)b * TOPK + s) * H + t]);
  out[(long)b * H + t] = acc / sum;
}

// ===================== host orchestration =====================
extern "C" void kernel_launch(void* const* d_in, const int* in_sizes, int n_in,
                              void* d_out, int out_size, void* d_ws, size_t ws_size,
                              hipStream_t stream) {
  (void)in_sizes; (void)n_in; (void)out_size; (void)ws_size;
  const float* ent_emb  = (const float*)d_in[0];
  const float* rel_emb  = (const float*)d_in[1];
  const float* wn_all   = (const float*)d_in[2];
  const float* ws_all   = (const float*)d_in[3];
  const float* lin_w    = (const float*)d_in[4];
  const float* lin_b    = (const float*)d_in[5];
  const float* gru_wih  = (const float*)d_in[6];
  const float* gru_whh  = (const float*)d_in[7];
  const float* gru_bih  = (const float*)d_in[8];
  const float* gru_bhh  = (const float*)d_in[9];
  const float* oc_w     = (const float*)d_in[10];
  const float* oc_b     = (const float*)d_in[11];
  const float* ofc_w    = (const float*)d_in[12];
  const float* ofc_b    = (const float*)d_in[13];
  const float* rc_w     = (const float*)d_in[14];
  const float* rc_b     = (const float*)d_in[15];
  const float* rfc_w    = (const float*)d_in[16];
  const float* rfc_b    = (const float*)d_in[17];
  const float* lin2_w   = (const float*)d_in[18];
  const float* lin2_b   = (const float*)d_in[19];
  const float* wq = (const float*)d_in[20];
  const float* wk = (const float*)d_in[21];
  const float* wv = (const float*)d_in[22];
  const float* wo = (const float*)d_in[23];
  const float* bqv = (const float*)d_in[24];
  const float* bk = (const float*)d_in[25];
  const float* bv = (const float*)d_in[26];
  const float* bo = (const float*)d_in[27];
  const int* src = (const int*)d_in[28];
  const int* dst = (const int*)d_in[29];
  const int* rid = (const int*)d_in[30];
  const int* subj = (const int*)d_in[31];
  const int* rel = (const int*)d_in[32];
  const int* obj = (const int*)d_in[33];

  // ---- workspace bump allocator (peak ~264 MB) ----
  char* wsb = (char*)d_ws;
  size_t off = 0;
  auto alloc = [&](size_t bytes) -> char* {
    char* p = wsb + off; off = (off + bytes + 255) & ~(size_t)255; return p;
  };
  float*    entCur  = (float*)alloc((size_t)N_ENT * H * 4);
  float*    relBuf0 = (float*)alloc((size_t)N_REL * H * 4);
  float*    relBuf1 = (float*)alloc((size_t)N_REL * H * 4);
  float*    gruIn   = (float*)alloc((size_t)N_REL * 2 * H * 4);
  float*    gi      = (float*)alloc((size_t)N_REL * 3 * H * 4);
  float*    gh      = (float*)alloc((size_t)N_REL * 3 * H * 4);
  float*    relEnt  = (float*)alloc((size_t)N_REL * H * 4);
  unsigned* maskb   = (unsigned*)alloc((size_t)N_REL * MASKW * 4);
  float*    cnt     = (float*)alloc((size_t)N_ENT * 4);
  size_t scanBase = off;
  float* agg = (float*)alloc((size_t)N_ENT * H * 4);
  float* h0  = (float*)alloc((size_t)N_ENT * H * 4);
  float* h1  = (float*)alloc((size_t)N_ENT * H * 4);
  float* tmp = (float*)alloc((size_t)N_ENT * H * 4);
  float* uBuf = tmp;  // u reuses tmp (tmp dead before u is produced)
  // post-scan phase reuses the scan arena
  off = scanBase;
  float* es     = (float*)alloc((size_t)Bq * H * 4);
  float* er     = (float*)alloc((size_t)Bq * H * 4);
  float* eo     = (float*)alloc((size_t)Bq * H * 4);
  float* cat2   = (float*)alloc((size_t)Bq * 2 * H * 4);
  float* feat   = (float*)alloc((size_t)Bq * H * 4);
  float* qbuf   = (float*)alloc((size_t)Bq * H * 4);
  float* qproj  = (float*)alloc((size_t)Bq * H * 4);
  float* attout = (float*)alloc((size_t)Bq * H * 4);
  float* rrbuf  = (float*)alloc((size_t)Bq * H * 4);
  int*   topkI  = (int*)alloc((size_t)Bq * TOPK * 4);
  unsigned short* convY = (unsigned short*)alloc((size_t)Bq * CH * H * 2);
  float* logits1 = (float*)alloc((size_t)Bq * N_ENT * 4);
  // kh/vh (bf16) alias the logits1 region (logits1 is dead after topk)
  unsigned short* khB = (unsigned short*)logits1;
  unsigned short* vhB = khB + (size_t)Bq * TOPK * H;

  // four used configurations:
  //   plainT : A f32, B transposed, C f32      (x @ W.T, x @ ent.T, ...)
  //   plainN : A f32, B plain KxN, C f32       (RGCN weight GEMMs)
  //   bf16A  : A bf16, B transposed, C f32     (ConvTransE FC)
  //   gather : A f32 row-gather, B trans, C bf16 (k/v projections over topk)
  auto gemmT = [&](const float* A, const float* Bm, const float* addC, const float* bias,
                   float* C, int M, int N, int K, int act) {
    dim3 g((N + 63) / 64, (M + 63) / 64);
    gemm_wmma_kernel<0, 0, 1, 0><<<g, 256, 0, stream>>>(A, nullptr, Bm, addC, bias, C, M, N, K, act);
  };
  auto gemmN = [&](const float* A, const float* Bm, const float* addC, const float* bias,
                   float* C, int M, int N, int K, int act) {
    dim3 g((N + 63) / 64, (M + 63) / 64);
    gemm_wmma_kernel<0, 0, 0, 0><<<g, 256, 0, stream>>>(A, nullptr, Bm, addC, bias, C, M, N, K, act);
  };
  auto gemmBF = [&](const unsigned short* A, const float* Bm, const float* bias,
                    float* C, int M, int N, int K, int act) {
    dim3 g((N + 63) / 64, (M + 63) / 64);
    gemm_wmma_kernel<1, 0, 1, 0><<<g, 256, 0, stream>>>(A, nullptr, Bm, nullptr, bias, C, M, N, K, act);
  };
  auto gemmG = [&](const float* A, const int* rowIdx, const float* Bm, const float* bias,
                   unsigned short* C, int M, int N, int K) {
    dim3 g((N + 63) / 64, (M + 63) / 64);
    gemm_wmma_kernel<0, 1, 1, 1><<<g, 256, 0, stream>>>(A, rowIdx, Bm, nullptr, bias, C, M, N, K, 0);
  };
  auto zf = [&](float* p, long n) {
    zero_kernel<<<(int)((n + 255) / 256), 256, 0, stream>>>(p, n);
  };

  // init: l2-normalized embeddings
  rownorm_kernel<<<N_ENT, 256, 0, stream>>>(ent_emb, entCur);
  rownorm_kernel<<<N_REL, 256, 0, stream>>>(rel_emb, relBuf0);
  float* relCur = relBuf0;
  float* relNxt = relBuf1;

  // -------- temporal scan --------
  for (int t = 0; t < T; ++t) {
    const int* sT = src + (long)t * E;
    const int* dT = dst + (long)t * E;
    const int* rT = rid + (long)t * E;
    // rel-node aggregation with dedup mask
    zf((float*)maskb, (long)N_REL * MASKW);
    mask_set_kernel<<<(E + 255) / 256, 256, 0, stream>>>(sT, dT, rT, maskb);
    rel_agg_kernel<<<N_REL, 256, 0, stream>>>(maskb, entCur, relEnt);
    // GRU: gi = [relEnt, rel_emb_raw] @ Wih.T + bih ; gh = relCur @ Whh.T + bhh
    gru_in_kernel<<<N_REL, 256, 0, stream>>>(relEnt, rel_emb, gruIn);
    gemmT(gruIn, gru_wih, nullptr, gru_bih, gi, N_REL, 3 * H, 2 * H, 0);
    gemmT(relCur, gru_whh, nullptr, gru_bhh, gh, N_REL, 3 * H, H, 0);
    gru_fuse_kernel<<<N_REL, 256, 0, stream>>>(gi, gh, relCur, relNxt);
    // RGCN layers
    const float* hp = entCur;
    float* hb[2] = { h0, h1 };
    for (int l = 0; l < L; ++l) {
      zf(agg, (long)N_ENT * H);
      zf(cnt, N_ENT);
      scatter_kernel<<<(E * 64 + 255) / 256, 256, 0, stream>>>(sT, dT, rT, hp, relNxt, agg, cnt);
      agg_norm_kernel<<<(int)(((long)N_ENT * H + 255) / 256), 256, 0, stream>>>(agg, cnt);
      gemmN(agg, wn_all + (size_t)l * H * H, nullptr, nullptr, tmp, N_ENT, H, H, 0);
      gemmN(hp, ws_all + (size_t)l * H * H, tmp, nullptr, hb[l], N_ENT, H, H, 1);
      hp = hb[l];
    }
    rownorm_kernel<<<N_ENT, 256, 0, stream>>>(h1, h1);
    // gated update: u = sigmoid(ent @ lin_w.T + lin_b); ent += u*(h-ent)
    gemmT(entCur, lin_w, nullptr, lin_b, uBuf, N_ENT, H, H, 2);
    ent_update_kernel<<<(int)(((long)N_ENT * H + 255) / 256), 256, 0, stream>>>(entCur, uBuf, h1);
    // swap rel buffers
    float* tmpp = relCur; relCur = relNxt; relNxt = tmpp;
  }

  // -------- decoding --------
  gather_kernel<<<Bq, 256, 0, stream>>>(entCur, relCur, subj, rel, obj, es, er, eo, cat2);

  // object branch: ConvTransE -> logits -> topk
  conv_feat_kernel<<<Bq, 256, 0, stream>>>(es, er, oc_w, oc_b, convY);
  gemmBF(convY, ofc_w, ofc_b, feat, Bq, H, CH * H, 1);
  gemmT(feat, entCur, nullptr, nullptr, logits1, Bq, N_ENT, H, 0);
  topk_kernel<<<Bq, 256, 0, stream>>>(logits1, topkI);

  // attention refinement
  gemmT(cat2, lin2_w, nullptr, lin2_b, qbuf, Bq, H, 2 * H, 0);
  gemmT(qbuf, wq, nullptr, bqv, qproj, Bq, H, H, 0);
  gemmG(entCur, topkI, wk, bk, khB, Bq * TOPK, H, H);
  gemmG(entCur, topkI, wv, bv, vhB, Bq * TOPK, H, H);
  attn_kernel<<<Bq, 256, 0, stream>>>(qproj, khB, vhB, attout);
  gemmT(attout, wo, nullptr, bo, rrbuf, Bq, H, H, 0);

  float* outObj = (float*)d_out;
  float* outRel = outObj + (long)Bq * N_ENT;
  gemmT(rrbuf, entCur, nullptr, nullptr, outObj, Bq, N_ENT, H, 0);

  // relation branch
  conv_feat_kernel<<<Bq, 256, 0, stream>>>(es, eo, rc_w, rc_b, convY);
  gemmBF(convY, rfc_w, rfc_b, feat, Bq, H, CH * H, 1);
  gemmT(feat, relCur, nullptr, nullptr, outRel, Bq, N_REL, H, 0);
}
